// RecursiveMamba130M_43250320670933
// MI455X (gfx1250) — matrix operand order repaired and checked by reading, not compiled, not run on hardware
//
#include <hip/hip_runtime.h>
#include <hip/hip_bf16.h>

// ---------------- model constants ----------------
#define D_MODEL   768
#define N_LAYER   24
#define N_LOWER   22
#define N_REASON  3
#define D_INNER   1536
#define D_STATE   16
#define D_CONV    4
#define DT_RANK   48
#define SEQ_L     1024
#define VOCAB     50280
#define EPS       1e-5f

// ---------------- vector types ----------------
typedef __bf16  bf16x16 __attribute__((ext_vector_type(16)));
typedef float   f32x8   __attribute__((ext_vector_type(8)));
typedef unsigned int u32x4 __attribute__((ext_vector_type(4)));
typedef int     i32x8   __attribute__((ext_vector_type(8)));
typedef int     i32x4   __attribute__((ext_vector_type(4)));

union Frag { bf16x16 v; u32x4 q[2]; unsigned short u[16]; };

__device__ __forceinline__ unsigned short f2bf(float f) {
    unsigned int u = __float_as_uint(f);
    unsigned int r = u + 0x7FFFu + ((u >> 16) & 1u);   // round-to-nearest-even
    return (unsigned short)(r >> 16);
}

// ---------------- workspace layout (bytes, all 256-aligned) ----------------
#define OFF_INW   ((size_t)0)            // bf16 in_proj_w   24*3072*768
#define OFF_OUTW  ((size_t)113246208)    // bf16 out_proj_w  24*768*1536
#define OFF_LMW   ((size_t)169869312)    // bf16 lm_head_w   50280*768
#define OFF_X     ((size_t)247099392)    // f32 x            1024*768
#define OFF_RES   ((size_t)250245120)    // f32 res          1024*768
#define OFF_LRES  ((size_t)253390848)    // f32 loop_res     1024*768
#define OFF_HBF   ((size_t)256536576)    // bf16 h           1024*768
#define OFF_XZ    ((size_t)258109440)    // f32 xz           1024*3072
#define OFF_XI    ((size_t)270692352)    // f32 xi           1024*1536
#define OFF_DBC   ((size_t)276983808)    // f32 dbc          1024*80
#define OFF_DT    ((size_t)277311488)    // f32 dt           1024*1536
#define OFF_Y     ((size_t)283602944)    // f32 y(scan)      1024*1536
#define OFF_YBF   ((size_t)289894400)    // bf16 gated y     1024*1536
#define OFF_XFBF  ((size_t)293040128)    // bf16 final x     1024*768

// ================= kernels =================

__global__ void k_cvt_bf16(const float* __restrict__ s, unsigned short* __restrict__ d, long n) {
    long i = (long)blockIdx.x * blockDim.x + threadIdx.x;
    long stride = (long)gridDim.x * blockDim.x;
    for (; i < n; i += stride) d[i] = f2bf(s[i]);
}

__global__ void k_embed(const int* __restrict__ ids, const float* __restrict__ emb,
                        float* __restrict__ x, float* __restrict__ res) {
    int i = blockIdx.x * blockDim.x + threadIdx.x;
    if (i >= SEQ_L * D_MODEL) return;
    int l = i / D_MODEL, c = i % D_MODEL;
    x[i] = emb[(size_t)ids[l] * D_MODEL + c];
    res[i] = 0.f;
}

// res = x + res;  h_bf16 = rmsnorm(res, w)
__global__ void __launch_bounds__(256)
k_prenorm(const float* __restrict__ x, float* __restrict__ res,
          const float* __restrict__ w, unsigned short* __restrict__ hbf) {
    __shared__ float sred[256];
    const int l = blockIdx.x, t = threadIdx.x;
    float s = 0.f;
    for (int c = t; c < D_MODEL; c += 256) {
        float v = x[l * D_MODEL + c] + res[l * D_MODEL + c];
        res[l * D_MODEL + c] = v;
        s += v * v;
    }
    sred[t] = s; __syncthreads();
    for (int off = 128; off > 0; off >>= 1) {
        if (t < off) sred[t] += sred[t + off];
        __syncthreads();
    }
    float rms = rsqrtf(sred[0] / (float)D_MODEL + EPS);
    for (int c = t; c < D_MODEL; c += 256)
        hbf[l * D_MODEL + c] = f2bf(res[l * D_MODEL + c] * rms * w[c]);
}

// xf_bf16 = rmsnorm(x + res, w)
__global__ void __launch_bounds__(256)
k_finalnorm(const float* __restrict__ x, const float* __restrict__ res,
            const float* __restrict__ w, unsigned short* __restrict__ out) {
    __shared__ float sred[256];
    const int l = blockIdx.x, t = threadIdx.x;
    float s = 0.f;
    for (int c = t; c < D_MODEL; c += 256) {
        float v = x[l * D_MODEL + c] + res[l * D_MODEL + c];
        s += v * v;
    }
    sred[t] = s; __syncthreads();
    for (int off = 128; off > 0; off >>= 1) {
        if (t < off) sred[t] += sred[t + off];
        __syncthreads();
    }
    float rms = rsqrtf(sred[0] / (float)D_MODEL + EPS);
    for (int c = t; c < D_MODEL; c += 256) {
        float v = x[l * D_MODEL + c] + res[l * D_MODEL + c];
        out[l * D_MODEL + c] = f2bf(v * rms * w[c]);
    }
}

__device__ __forceinline__ void load_b4(Frag* b,
                                        const unsigned short* wp0, const unsigned short* wp1,
                                        const unsigned short* wp2, const unsigned short* wp3,
                                        int k) {
    b[0].q[0] = *(const u32x4*)(wp0 + k);  b[0].q[1] = *(const u32x4*)(wp0 + k + 8);
    b[1].q[0] = *(const u32x4*)(wp1 + k);  b[1].q[1] = *(const u32x4*)(wp1 + k + 8);
    b[2].q[0] = *(const u32x4*)(wp2 + k);  b[2].q[1] = *(const u32x4*)(wp2 + k + 8);
    b[3].q[0] = *(const u32x4*)(wp3 + k);  b[3].q[1] = *(const u32x4*)(wp3 + k + 8);
}

__device__ __forceinline__ void wmma4(const Frag& a, const Frag* b, f32x8* acc) {
    acc[0] = __builtin_amdgcn_wmma_f32_16x16x32_bf16(false, a.v, false, b[0].v, (short)0, acc[0], false, false);
    acc[1] = __builtin_amdgcn_wmma_f32_16x16x32_bf16(false, a.v, false, b[1].v, (short)0, acc[1], false, false);
    acc[2] = __builtin_amdgcn_wmma_f32_16x16x32_bf16(false, a.v, false, b[2].v, (short)0, acc[2], false, false);
    acc[3] = __builtin_amdgcn_wmma_f32_16x16x32_bf16(false, a.v, false, b[3].v, (short)0, acc[3], false, false);
}

// Y(MxN,f32) = Xbf16(MxK,row) * Wbf16(NxK,row)^T  via v_wmma_f32_16x16x32_bf16.
// A tile (16 x K) staged into LDS by the Tensor Data Mover (one D# per row,
// +8-element row pad for bank spread); B fragments ping-pong double-buffered
// from global with an unroll-by-2 loop (no register rotation copies).
// block = 8 waves; wave tile 16(M) x 64(N); block tile 16 x 512; N edge-guarded.
__global__ void __launch_bounds__(256)
k_gemm_bf16(const unsigned short* __restrict__ X, const unsigned short* __restrict__ W,
            float* __restrict__ Y, int M, int N, int K) {
    extern __shared__ unsigned short smA[];           // 16 x (K+8) bf16
    const int lane = threadIdx.x & 31;
    const int wave = threadIdx.x >> 5;
    const int m0 = blockIdx.y << 4;
    const int n0 = blockIdx.x * 512 + wave * 64;
    const int ldaP = K + 8;                           // padded LDS row (elements)

    // ---- TDM: DMA the 16-row A tile into LDS (wave 0 issues; EXEC ignored) ----
    if (threadIdx.x < 32) {
        unsigned ldsBase = (unsigned)(unsigned long long)(void*)smA;  // LDS byte offset
        for (int r = 0; r < 16; ++r) {
            unsigned long long ga =
                (unsigned long long)(const void*)(X + (size_t)(m0 + r) * K);
            u32x4 g0; i32x8 g1;
            i32x4 g2 = {0,0,0,0}, g3 = {0,0,0,0};
            i32x8 g4 = {0,0,0,0,0,0,0,0};
            // D# group0: count=1 | lds_addr | global_addr[56:0] | type=2
            g0[0] = 1u;
            g0[1] = ldsBase + (unsigned)(r * ldaP * 2);
            g0[2] = (unsigned)(ga & 0xFFFFFFFFull);
            g0[3] = (unsigned)((ga >> 32) & 0x01FFFFFFull) | 0x80000000u;
            // D# group1: data_size=2B; tensor_dim0=K, tensor_dim1=1;
            //            tile_dim0=K, tile_dim1=1; dim0_stride=K
            unsigned td0 = (unsigned)K;
            g1[0] = (int)(1u << 16);                            // data_size code 1 (2 bytes)
            g1[1] = (int)((td0 & 0xFFFFu) << 16);               // tensor_dim0[15:0]
            g1[2] = (int)(((td0 >> 16) & 0xFFFFu) | (1u << 16));// td0[31:16] | tensor_dim1=1
            g1[3] = (int)((td0 & 0xFFFFu) << 16);               // td1 hi=0 | tile_dim0=K
            g1[4] = 1;                                          // tile_dim1=1, tile_dim2=0
            g1[5] = (int)td0;                                   // tensor_dim0_stride lo
            g1[6] = 0;                                          // stride hi / dim1_stride lo
            g1[7] = 0;
            __builtin_amdgcn_tensor_load_to_lds(g0, g1, g2, g3, g4, 0);
        }
        __builtin_amdgcn_s_wait_tensorcnt(0);
    }
    __syncthreads();

    f32x8 acc[4];
    acc[0] = (f32x8){0.f,0.f,0.f,0.f,0.f,0.f,0.f,0.f};
    acc[1] = acc[0]; acc[2] = acc[0]; acc[3] = acc[0];

    const int kA = (lane >> 4) << 4;   // A: 16 contiguous K values
    const int kB = (lane >> 4) << 4;   // B: 16 contiguous K values
    const unsigned short* aRow = smA + (lane & 15) * ldaP;

    int nr0 = n0 +  0 + (lane & 15); if (nr0 > N - 1) nr0 = N - 1;
    int nr1 = n0 + 16 + (lane & 15); if (nr1 > N - 1) nr1 = N - 1;
    int nr2 = n0 + 32 + (lane & 15); if (nr2 > N - 1) nr2 = N - 1;
    int nr3 = n0 + 48 + (lane & 15); if (nr3 > N - 1) nr3 = N - 1;

    const unsigned short* wp0 = W + (size_t)nr0 * K + kB;
    const unsigned short* wp1 = W + (size_t)nr1 * K + kB;
    const unsigned short* wp2 = W + (size_t)nr2 * K + kB;
    const unsigned short* wp3 = W + (size_t)nr3 * K + kB;

    // ping-pong B buffers; K is a multiple of 64 for all GEMMs in this model
    Frag bb0[4], bb1[4];
    load_b4(bb0, wp0, wp1, wp2, wp3, 0);

    for (int k = 0; k < K; k += 64) {
        Frag a0;                                      // ds_load_b128 x2 (bank-padded)
        a0.q[0] = *(const u32x4*)(aRow + k + kA);
        a0.q[1] = *(const u32x4*)(aRow + k + 8 + kA);
        if (k + 32 < K) load_b4(bb1, wp0, wp1, wp2, wp3, k + 32);
        __builtin_prefetch(wp0 + k + 256, 0, 0);
        wmma4(a0, bb0, acc);

        if (k + 32 < K) {
            Frag a1;
            a1.q[0] = *(const u32x4*)(aRow + k + 32 + kA);
            a1.q[1] = *(const u32x4*)(aRow + k + 40 + kA);
            if (k + 64 < K) load_b4(bb0, wp0, wp1, wp2, wp3, k + 64);
            wmma4(a1, bb1, acc);
        }
    }

    // D layout: lane = col(n) + 16*(m>=8); vgpr r -> row (lane>>4)*8 + r
    const int ns = lane & 15;
    const int ms = m0 + ((lane >> 4) << 3);
    #pragma unroll
    for (int r = 0; r < 8; ++r) {
        size_t row = (size_t)(ms + r) * N;
        int c0 = n0 +  0 + ns; if (c0 < N) Y[row + c0] = acc[0][r];
        int c1 = n0 + 16 + ns; if (c1 < N) Y[row + c1] = acc[1][r];
        int c2 = n0 + 32 + ns; if (c2 < N) Y[row + c2] = acc[2][r];
        int c3 = n0 + 48 + ns; if (c3 < N) Y[row + c3] = acc[3][r];
    }
}

// depthwise causal conv (K=4) over xi-part of xz, + bias, + SiLU
__global__ void k_conv(const float* __restrict__ xz, const float* __restrict__ cw,
                       const float* __restrict__ cb, float* __restrict__ xi) {
    int i = blockIdx.x * blockDim.x + threadIdx.x;
    if (i >= SEQ_L * D_INNER) return;
    int l = i / D_INNER, d = i % D_INNER;
    float acc = cb[d];
    #pragma unroll
    for (int j = 0; j < D_CONV; ++j) {
        int ls = l - (D_CONV - 1) + j;
        if (ls >= 0) acc += cw[d * D_CONV + j] * xz[ls * (2 * D_INNER) + d];
    }
    xi[i] = acc / (1.f + __expf(-acc));
}

// dbc(1024x80) = xi(1024x1536) @ x_proj_w(80x1536)^T  (exact fp32; tiny GEMM)
__global__ void __launch_bounds__(128)
k_xproj(const float* __restrict__ xi, const float* __restrict__ xw, float* __restrict__ dbc) {
    int l = blockIdx.x, e = threadIdx.x;
    if (e >= DT_RANK + 2 * D_STATE) return;
    const float* xr = xi + (size_t)l * D_INNER;
    const float* wr = xw + (size_t)e * D_INNER;
    float s = 0.f;
    for (int k = 0; k < D_INNER; ++k) s += xr[k] * wr[k];
    dbc[l * (DT_RANK + 2 * D_STATE) + e] = s;
}

// dt(1024x1536) = softplus(dbc[:, :48] @ dt_proj_w^T + b)  (exact fp32)
__global__ void k_dtproj(const float* __restrict__ dbc, const float* __restrict__ dtw,
                         const float* __restrict__ dtb, float* __restrict__ dt) {
    int i = blockIdx.x * blockDim.x + threadIdx.x;
    if (i >= SEQ_L * D_INNER) return;
    int l = i / D_INNER, d = i % D_INNER;
    const float* r = dbc + l * (DT_RANK + 2 * D_STATE);
    const float* w = dtw + (size_t)d * DT_RANK;
    float s = dtb[d];
    for (int k = 0; k < DT_RANK; ++k) s += r[k] * w[k];
    dt[i] = (s > 15.f) ? s : log1pf(__expf(s));
}

// selective scan: thread per (d,n); 16-lane shuffle reduction over n (wave32)
__global__ void __launch_bounds__(256)
k_scan(const float* __restrict__ dt, const float* __restrict__ dbc,
       const float* __restrict__ xi, const float* __restrict__ A_log,
       float* __restrict__ y) {
    const int t = threadIdx.x;
    const int n = t & 15;
    const int d = blockIdx.x * 16 + (t >> 4);
    const float a = -__expf(A_log[d * D_STATE + n]);
    float h = 0.f;
    for (int l = 0; l < SEQ_L; ++l) {
        float dtv = dt[l * D_INNER + d];
        float xv  = xi[l * D_INNER + d];
        float bv  = dbc[l * (DT_RANK + 2 * D_STATE) + DT_RANK + n];
        float cv  = dbc[l * (DT_RANK + 2 * D_STATE) + DT_RANK + D_STATE + n];
        h = __expf(dtv * a) * h + dtv * bv * xv;
        float p = h * cv;
        p += __shfl_xor(p, 1);
        p += __shfl_xor(p, 2);
        p += __shfl_xor(p, 4);
        p += __shfl_xor(p, 8);
        if (n == 0) y[l * D_INNER + d] = p;
    }
}

// y = (y_scan + D*xi) * silu(z) -> bf16
__global__ void k_gate(const float* __restrict__ y, const float* __restrict__ xi,
                       const float* __restrict__ xz, const float* __restrict__ Dp,
                       unsigned short* __restrict__ ybf) {
    int i = blockIdx.x * blockDim.x + threadIdx.x;
    if (i >= SEQ_L * D_INNER) return;
    int l = i / D_INNER, d = i % D_INNER;
    float v = y[i] + Dp[d] * xi[i];
    float z = xz[l * (2 * D_INNER) + D_INNER + d];
    v *= z / (1.f + __expf(-z));
    ybf[i] = f2bf(v);
}

__global__ void k_copy(float* __restrict__ dst, const float* __restrict__ src, int n) {
    int i = blockIdx.x * blockDim.x + threadIdx.x;
    if (i < n) dst[i] = src[i];
}
__global__ void k_axpy(float* __restrict__ res, const float* __restrict__ lres, int n) {
    int i = blockIdx.x * blockDim.x + threadIdx.x;
    if (i < n) res[i] += lres[i];
}

// ================= host orchestration =================
extern "C" void kernel_launch(void* const* d_in, const int* in_sizes, int n_in,
                              void* d_out, int out_size, void* d_ws, size_t ws_size,
                              hipStream_t stream) {
    (void)in_sizes; (void)n_in; (void)out_size; (void)ws_size;

    const int*   ids      = (const int*)  d_in[0];
    const float* emb      = (const float*)d_in[1];
    const float* in_w     = (const float*)d_in[2];
    const float* conv_w   = (const float*)d_in[3];
    const float* conv_b   = (const float*)d_in[4];
    const float* xproj_w  = (const float*)d_in[5];
    const float* dtproj_w = (const float*)d_in[6];
    const float* dtproj_b = (const float*)d_in[7];
    const float* A_log    = (const float*)d_in[8];
    const float* Dp       = (const float*)d_in[9];
    const float* out_w    = (const float*)d_in[10];
    const float* norm_w   = (const float*)d_in[11];
    const float* norm_f_w = (const float*)d_in[12];
    const float* lm_w     = (const float*)d_in[13];

    char* ws = (char*)d_ws;
    unsigned short* INW  = (unsigned short*)(ws + OFF_INW);
    unsigned short* OUTW = (unsigned short*)(ws + OFF_OUTW);
    unsigned short* LMW  = (unsigned short*)(ws + OFF_LMW);
    float* X    = (float*)(ws + OFF_X);
    float* RES  = (float*)(ws + OFF_RES);
    float* LRES = (float*)(ws + OFF_LRES);
    unsigned short* HBF = (unsigned short*)(ws + OFF_HBF);
    float* XZ   = (float*)(ws + OFF_XZ);
    float* XI   = (float*)(ws + OFF_XI);
    float* DBC  = (float*)(ws + OFF_DBC);
    float* DT   = (float*)(ws + OFF_DT);
    float* Yb   = (float*)(ws + OFF_Y);
    unsigned short* YBF  = (unsigned short*)(ws + OFF_YBF);
    unsigned short* XFBF = (unsigned short*)(ws + OFF_XFBF);

    // one-time (per call) bf16 weight conversion
    k_cvt_bf16<<<4096, 256, 0, stream>>>(in_w,  INW,  (long)N_LAYER * 2 * D_INNER * D_MODEL);
    k_cvt_bf16<<<4096, 256, 0, stream>>>(out_w, OUTW, (long)N_LAYER * D_MODEL * D_INNER);
    k_cvt_bf16<<<4096, 256, 0, stream>>>(lm_w,  LMW,  (long)VOCAB * D_MODEL);

    k_embed<<<(SEQ_L * D_MODEL + 255) / 256, 256, 0, stream>>>(ids, emb, X, RES);

    const int nLD = SEQ_L * D_INNER;
    const int gLD = (nLD + 255) / 256;
    const int nLM = SEQ_L * D_MODEL;
    const int gLM = (nLM + 255) / 256;

    const size_t smK768  = (size_t)16 * (D_MODEL + 8) * 2;   // A-tile LDS, K=768
    const size_t smK1536 = (size_t)16 * (D_INNER + 8) * 2;   // A-tile LDS, K=1536

    auto layer_step = [&](int layer) {
        k_prenorm<<<SEQ_L, 256, 0, stream>>>(X, RES, norm_w + (size_t)layer * D_MODEL, HBF);
        dim3 g1((2 * D_INNER + 511) / 512, SEQ_L / 16);
        k_gemm_bf16<<<g1, 256, smK768, stream>>>(HBF, INW + (size_t)layer * 2 * D_INNER * D_MODEL,
                                                 XZ, SEQ_L, 2 * D_INNER, D_MODEL);
        k_conv<<<gLD, 256, 0, stream>>>(XZ, conv_w + (size_t)layer * D_INNER * D_CONV,
                                        conv_b + (size_t)layer * D_INNER, XI);
        k_xproj<<<SEQ_L, 128, 0, stream>>>(XI, xproj_w + (size_t)layer * (DT_RANK + 2 * D_STATE) * D_INNER, DBC);
        k_dtproj<<<gLD, 256, 0, stream>>>(DBC, dtproj_w + (size_t)layer * D_INNER * DT_RANK,
                                          dtproj_b + (size_t)layer * D_INNER, DT);
        k_scan<<<D_INNER / 16, 256, 0, stream>>>(DT, DBC, XI, A_log + (size_t)layer * D_INNER * D_STATE, Yb);
        k_gate<<<gLD, 256, 0, stream>>>(Yb, XI, XZ, Dp + (size_t)layer * D_INNER, YBF);
        dim3 g2((D_MODEL + 511) / 512, SEQ_L / 16);
        k_gemm_bf16<<<g2, 256, smK1536, stream>>>(YBF, OUTW + (size_t)layer * D_MODEL * D_INNER,
                                                  X, SEQ_L, D_MODEL, D_INNER);
    };

    for (int layer = 0; layer < N_LOWER; ++layer) layer_step(layer);

    for (int r = 0; r < N_REASON; ++r) {
        k_copy<<<gLM, 256, 0, stream>>>(LRES, RES, nLM);
        for (int layer = N_LOWER; layer < N_LAYER; ++layer) layer_step(layer);
        k_axpy<<<gLM, 256, 0, stream>>>(RES, LRES, nLM);
    }

    k_finalnorm<<<SEQ_L, 256, 0, stream>>>(X, RES, norm_f_w, XFBF);
    dim3 g3((VOCAB + 511) / 512, SEQ_L / 16);
    k_gemm_bf16<<<g3, 256, smK768, stream>>>(XFBF, LMW, (float*)d_out, SEQ_L, VOCAB, D_MODEL);
}
